// MultiScaleRetention_80290118631463
// MI455X (gfx1250) — matrix operand order repaired and checked by reading, not compiled, not run on hardware
//
#include <hip/hip_runtime.h>
#include <math.h>

// ---------------- problem constants ----------------
#define BATCH 2
#define LSEQ  2048
#define HID   1024
#define NHEAD 8
#define DQK   128
#define DV    256
#define VDIM  2048
#define MROWS (BATCH * LSEQ)          // 4096
#define QSCALE 0.08838834764831845f   // 1/sqrt(128)
#define LOG2_10000 13.287712379549449f

typedef __attribute__((ext_vector_type(16))) __bf16 v16bf;
typedef __attribute__((ext_vector_type(8)))  float  v8f;

__device__ __forceinline__ v8f wmma_bf16(v16bf a, v16bf b, v8f c) {
  // D = A(16x32) * B(32x16) + C, fp32 accumulate
  return __builtin_amdgcn_wmma_f32_16x16x32_bf16(
      /*neg_a=*/false, a, /*neg_b=*/false, b,
      /*c_mod=*/(short)0, c, /*reuse_a=*/false, /*reuse_b=*/false);
}

// A fragment (16x32 bf16): lane m=lane&15 holds row m; lane half selects K
// K=0..15 (half 0) vs K=16..31 (half 1).
__device__ __forceinline__ v16bf load_frag_a(const __bf16* base, int ld,
                                             int row, int k, int half) {
  union { uint4 u[2]; v16bf v; } f;
  const __bf16* p = base + (size_t)row * ld + k + half * 16;
  f.u[0] = *(const uint4*)(p);
  f.u[1] = *(const uint4*)(p + 8);
  return f.v;
}

// B fragment (32x16 bf16) from row-major Bt[N][K] (i.e. B transposed):
// lane n=lane&15 holds column n; lanes 0-15 K=0..15, lanes 16-31 K=16..31.
__device__ __forceinline__ v16bf load_frag_b(const __bf16* base, int ld,
                                             int col, int k, int half) {
  union { uint4 u[2]; v16bf v; } f;
  const __bf16* p = base + (size_t)col * ld + k + half * 16;
  f.u[0] = *(const uint4*)(p);
  f.u[1] = *(const uint4*)(p + 8);
  return f.v;
}

// ---------------- elementwise converts ----------------
__global__ void k_f32_to_bf16(const float* __restrict__ in,
                              __bf16* __restrict__ out, int n) {
  int i = blockIdx.x * blockDim.x + threadIdx.x;
  if (i < n) out[i] = (__bf16)in[i];
}

// in: fp32 [K,N] row-major  ->  out: bf16 [N,K] row-major
__global__ void k_transpose_to_bf16(const float* __restrict__ in,
                                    __bf16* __restrict__ out, int K, int N) {
  int i = blockIdx.x * blockDim.x + threadIdx.x;
  if (i >= K * N) return;
  int k = i % K, n = i / K;
  out[i] = (__bf16)in[(size_t)k * N + n];
}

// ---------------- WMMA GEMM: C[M,N] = A[M,K] * Bt[N,K]^T ----------------
// One wave computes a 32x64 C tile. 8 WMMAs per 32-wide K chunk.
template <bool F32OUT>
__global__ void k_gemm_bf16(const __bf16* __restrict__ A,
                            const __bf16* __restrict__ Bt,
                            void* __restrict__ Cv, int M, int N, int K) {
  int wave = (blockIdx.x * blockDim.x + threadIdx.x) >> 5;
  int lane = threadIdx.x & 31;
  int wavesM = M >> 5, wavesN = N >> 6;
  if (wave >= wavesM * wavesN) return;
  int m0 = (wave % wavesM) * 32;
  int n0 = (wave / wavesM) * 64;
  int half = lane >> 4, l15 = lane & 15;

  v8f acc[2][4] = {};
  for (int k = 0; k < K; k += 32) {
    v16bf a0 = load_frag_a(A, K, m0 + l15, k, half);
    v16bf a1 = load_frag_a(A, K, m0 + 16 + l15, k, half);
#pragma unroll
    for (int j = 0; j < 4; ++j) {
      v16bf b = load_frag_b(Bt, K, n0 + 16 * j + l15, k, half);
      acc[0][j] = wmma_bf16(a0, b, acc[0][j]);
      acc[1][j] = wmma_bf16(a1, b, acc[1][j]);
    }
  }
#pragma unroll
  for (int i = 0; i < 2; ++i)
#pragma unroll
    for (int j = 0; j < 4; ++j)
#pragma unroll
      for (int r = 0; r < 8; ++r) {
        int row = m0 + 16 * i + half * 8 + r;
        int col = n0 + 16 * j + l15;
        if (F32OUT) ((float*)Cv)[(size_t)row * N + col] = acc[i][j][r];
        else        ((__bf16*)Cv)[(size_t)row * N + col] = (__bf16)acc[i][j][r];
      }
}

// ---------------- RoPE + repack [b,t,h,d] -> [b,h,t,d] ----------------
__global__ void k_rope_pack(const __bf16* __restrict__ qraw,
                            const __bf16* __restrict__ kraw,
                            __bf16* __restrict__ qb, __bf16* __restrict__ kb) {
  int idx = blockIdx.x * blockDim.x + threadIdx.x;  // B*L*NH*64 pairs
  if (idx >= BATCH * LSEQ * NHEAD * 64) return;
  int pair = idx & 63;
  int h = (idx >> 6) & (NHEAD - 1);
  int t = (idx >> 9) & (LSEQ - 1);
  int b = idx >> 20;
  float inv = exp2f(-((float)(2 * pair) / (float)DQK) * LOG2_10000);
  float ang = (float)t * inv;
  float c = cosf(ang), s = sinf(ang);
  size_t ib = (((size_t)b * LSEQ + t) * NHEAD + h) * DQK + 2 * pair;
  size_t ob = (((size_t)b * NHEAD + h) * LSEQ + t) * DQK + 2 * pair;
  float q0 = (float)qraw[ib], q1 = (float)qraw[ib + 1];
  qb[ob]     = (__bf16)(q0 * c - q1 * s);
  qb[ob + 1] = (__bf16)(q1 * c + q0 * s);
  float k0 = (float)kraw[ib], k1 = (float)kraw[ib + 1];
  kb[ob]     = (__bf16)(k0 * c - k1 * s);
  kb[ob + 1] = (__bf16)(k1 * c + k0 * s);
}

// ---------------- v [b,t,h,dv] -> vT [b,h,dv,t] ----------------
__global__ void k_pack_vt(const __bf16* __restrict__ vraw,
                          __bf16* __restrict__ vtb) {
  int idx = blockIdx.x * blockDim.x + threadIdx.x;  // B*L*NH*DV
  if (idx >= BATCH * LSEQ * NHEAD * DV) return;
  int dv = idx & (DV - 1);
  int h  = (idx >> 8) & (NHEAD - 1);
  int t  = (idx >> 11) & (LSEQ - 1);
  int b  = idx >> 22;
  vtb[(((size_t)b * NHEAD + h) * DV + dv) * LSEQ + t] = vraw[idx];
}

// ---------------- retention core ----------------
// wave = (b, h, 16-row query tile); full 16x256 fp32 accumulator per wave.
__global__ void __launch_bounds__(128)
k_retention(const __bf16* __restrict__ qb, const __bf16* __restrict__ kb,
            const __bf16* __restrict__ vtb, __bf16* __restrict__ ob) {
  int wave = (blockIdx.x * blockDim.x + threadIdx.x) >> 5;
  int lane = threadIdx.x & 31;
  int wblk = threadIdx.x >> 5;
  int mt = wave & 127;               // L/16 = 128 query tiles
  int h  = (wave >> 7) & (NHEAD - 1);
  int b  = wave >> 10;
  int m0 = mt * 16;
  int half = lane >> 4, l15 = lane & 15;

  float s = log2f(1.0f - exp2f(-5.0f - (float)h));   // per-head decay (log2)
  const __bf16* qh = qb  + ((size_t)b * NHEAD + h) * LSEQ * DQK;
  const __bf16* kh = kb  + ((size_t)b * NHEAD + h) * LSEQ * DQK;
  const __bf16* vh = vtb + ((size_t)b * NHEAD + h) * (size_t)DV * LSEQ;

  v16bf qf[4];
#pragma unroll
  for (int c = 0; c < 4; ++c) qf[c] = load_frag_a(qh, DQK, m0 + l15, c * 32, half);

  v8f acc[16] = {};
  __shared__ __attribute__((aligned(16))) __bf16 sS[4][16][32]; // [wave][row][key]

  int nkb = (m0 + 16 + 31) >> 5;     // causal: key blocks covering j <= i
  for (int kbi = 0; kbi < nkb; ++kbi) {
    int k0 = kbi * 32;
    v8f s0 = {}, s1 = {};
#pragma unroll
    for (int c = 0; c < 4; ++c) {
      v16bf kf0 = load_frag_b(kh, DQK, k0 + l15, c * 32, half);
      v16bf kf1 = load_frag_b(kh, DQK, k0 + 16 + l15, c * 32, half);
      s0 = wmma_bf16(qf[c], kf0, s0);
      s1 = wmma_bf16(qf[c], kf1, s1);
    }
    // decay * scale in C-layout (lane = key col, VGPR r = query row), then
    // stage to LDS to transpose into the A-fragment layout for S @ V.
#pragma unroll
    for (int r = 0; r < 8; ++r) {
      int i = m0 + half * 8 + r;
      int d0 = i - (k0 + l15);
      int d1 = d0 - 16;
      float w0 = (d0 >= 0) ? exp2f(s * (float)d0) * QSCALE : 0.0f;
      float w1 = (d1 >= 0) ? exp2f(s * (float)d1) * QSCALE : 0.0f;
      sS[wblk][half * 8 + r][l15]      = (__bf16)(s0[r] * w0);
      sS[wblk][half * 8 + r][16 + l15] = (__bf16)(s1[r] * w1);
    }
    // intra-wave LDS: DS ops execute in-order per wave, no barrier needed
    union { uint4 u[2]; v16bf v; } sa;
    const __bf16* sp = &sS[wblk][l15][0] + half * 16;
    sa.u[0] = *(const uint4*)(sp);
    sa.u[1] = *(const uint4*)(sp + 8);
    v16bf sfrag = sa.v;
#pragma unroll
    for (int t = 0; t < 16; ++t) {
      union { uint4 u[2]; v16bf v; } vf;
      const __bf16* vp = vh + (size_t)(t * 16 + l15) * LSEQ + k0 + half * 16;
      vf.u[0] = *(const uint4*)(vp);
      vf.u[1] = *(const uint4*)(vp + 8);
      acc[t] = wmma_bf16(sfrag, vf.v, acc[t]);
    }
  }
  // store o as bf16 [b, t, h, dv]
#pragma unroll
  for (int t = 0; t < 16; ++t)
#pragma unroll
    for (int r = 0; r < 8; ++r) {
      int row = m0 + half * 8 + r;
      ob[(((size_t)b * LSEQ + row) * NHEAD + h) * DV + t * 16 + l15] =
          (__bf16)acc[t][r];
    }
}

// ---------------- per-head RMSNorm + swish gate ----------------
__global__ void k_rmsnorm_gate(const __bf16* __restrict__ ob,
                               const __bf16* __restrict__ g,
                               const float* __restrict__ gw,
                               __bf16* __restrict__ out) {
  int wave = (blockIdx.x * blockDim.x + threadIdx.x) >> 5;  // row = (b*L+t)*NH+h
  int lane = threadIdx.x & 31;
  const __bf16* op = ob + (size_t)wave * DV;
  const __bf16* gp = g  + (size_t)wave * DV;
  union { uint4 u; __bf16 b[8]; } dv, gv, ov;
  dv.u = *(const uint4*)(op + lane * 8);
  gv.u = *(const uint4*)(gp + lane * 8);
  float v[8], ss = 0.0f;
#pragma unroll
  for (int e = 0; e < 8; ++e) { v[e] = (float)dv.b[e]; ss += v[e] * v[e]; }
#pragma unroll
  for (int m = 16; m >= 1; m >>= 1) ss += __shfl_xor(ss, m, 32);
  float r = rsqrtf(ss * (1.0f / DV) + 1e-5f);
#pragma unroll
  for (int e = 0; e < 8; ++e) {
    float gg = (float)gv.b[e];
    float gate = gg / (1.0f + expf(-gg));            // g * sigmoid(g)
    ov.b[e] = (__bf16)(v[e] * r * gw[lane * 8 + e] * gate);
  }
  *(uint4*)(((__bf16*)out) + (size_t)wave * DV + lane * 8) = ov.u;
}

// ---------------- launcher ----------------
extern "C" void kernel_launch(void* const* d_in, const int* in_sizes, int n_in,
                              void* d_out, int out_size, void* d_ws, size_t ws_size,
                              hipStream_t stream) {
  (void)in_sizes; (void)n_in; (void)out_size; (void)ws_size;
  const float* X  = (const float*)d_in[0];
  const float* Wq = (const float*)d_in[1];
  const float* Wk = (const float*)d_in[2];
  const float* Wv = (const float*)d_in[3];
  const float* Wg = (const float*)d_in[4];
  const float* Wo = (const float*)d_in[5];
  const float* gw = (const float*)d_in[6];

  const size_t MB = 1ull << 20;
  char* w = (char*)d_ws;
  __bf16* Xbf  = (__bf16*)(w + 0 * MB);    // 8 MB
  __bf16* WqT  = (__bf16*)(w + 8 * MB);    // 2 MB
  __bf16* WkT  = (__bf16*)(w + 10 * MB);   // 2 MB
  __bf16* WvT  = (__bf16*)(w + 12 * MB);   // 4 MB
  __bf16* WgT  = (__bf16*)(w + 16 * MB);   // 4 MB
  __bf16* WoT  = (__bf16*)(w + 20 * MB);   // 4 MB
  __bf16* qraw = (__bf16*)(w + 24 * MB);   // 8 MB
  __bf16* kraw = (__bf16*)(w + 32 * MB);   // 8 MB
  __bf16* vraw = (__bf16*)(w + 40 * MB);   // 16 MB
  __bf16* graw = (__bf16*)(w + 56 * MB);   // 16 MB
  __bf16* qb   = (__bf16*)(w + 72 * MB);   // 8 MB
  __bf16* kb   = (__bf16*)(w + 80 * MB);   // 8 MB
  __bf16* vtb  = (__bf16*)(w + 88 * MB);   // 16 MB  (total 104 MB)
  __bf16* ob   = qraw;                     // reuse: q/k raw dead after RoPE
  __bf16* obf  = vraw;                     // reuse: v raw dead after pack_vt

  // 1) converts
  {
    int n = MROWS * HID;
    k_f32_to_bf16<<<n / 256, 256, 0, stream>>>(X, Xbf, n);
  }
  k_transpose_to_bf16<<<(HID * HID) / 256, 256, 0, stream>>>(Wq, WqT, HID, HID);
  k_transpose_to_bf16<<<(HID * HID) / 256, 256, 0, stream>>>(Wk, WkT, HID, HID);
  k_transpose_to_bf16<<<(HID * VDIM) / 256, 256, 0, stream>>>(Wv, WvT, HID, VDIM);
  k_transpose_to_bf16<<<(HID * VDIM) / 256, 256, 0, stream>>>(Wg, WgT, HID, VDIM);
  k_transpose_to_bf16<<<(VDIM * HID) / 256, 256, 0, stream>>>(Wo, WoT, VDIM, HID);

  // 2) projection GEMMs (wave = 32x64 tile; 8 waves per 256-thread block)
  auto gemm_blocks = [](int M, int N) { return ((M / 32) * (N / 64)) / 8; };
  k_gemm_bf16<false><<<gemm_blocks(MROWS, HID), 256, 0, stream>>>(Xbf, WqT, qraw, MROWS, HID, HID);
  k_gemm_bf16<false><<<gemm_blocks(MROWS, HID), 256, 0, stream>>>(Xbf, WkT, kraw, MROWS, HID, HID);
  k_gemm_bf16<false><<<gemm_blocks(MROWS, VDIM), 256, 0, stream>>>(Xbf, WvT, vraw, MROWS, VDIM, HID);
  k_gemm_bf16<false><<<gemm_blocks(MROWS, VDIM), 256, 0, stream>>>(Xbf, WgT, graw, MROWS, VDIM, HID);

  // 3) RoPE + repack, v transpose
  k_rope_pack<<<(BATCH * LSEQ * NHEAD * 64) / 256, 256, 0, stream>>>(qraw, kraw, qb, kb);
  k_pack_vt<<<(BATCH * LSEQ * NHEAD * DV) / 256, 256, 0, stream>>>(vraw, vtb);

  // 4) retention: 2048 waves (b,h,16-row tile), 4 waves per block
  k_retention<<<(BATCH * NHEAD * (LSEQ / 16)) / 4, 128, 0, stream>>>(qb, kb, vtb, ob);

  // 5) RMSNorm + swish gate: one wave per (b,t,h) row
  k_rmsnorm_gate<<<(BATCH * LSEQ * NHEAD) / 8, 256, 0, stream>>>(ob, graw, gw, obf);

  // 6) output projection, fp32 epilogue into d_out
  k_gemm_bf16<true><<<gemm_blocks(MROWS, HID), 256, 0, stream>>>(obf, WoT, (float*)d_out, MROWS, HID, VDIM);
}
